// TemporalGraphAttentionNetwork_17540646437563
// MI455X (gfx1250) — matrix operand order repaired and checked by reading, not compiled, not run on hardware
//
#include <hip/hip_runtime.h>
#include <hip/hip_bf16.h>
#include <math.h>

typedef _Float16 f16_t;
typedef _Float16 v16h __attribute__((ext_vector_type(16)));
typedef _Float16 v8h  __attribute__((ext_vector_type(8)));
typedef float    v8f  __attribute__((ext_vector_type(8)));

#define B_TOT 16384
#define NNB   32
#define ND_   128
#define TD_   128
#define COMB_ 256
#define AD_   32
#define NHAD_ 128
#define FF_   512
#define L_    2

#define BT       16
#define NTHREADS 128
#define KSTR     136   // padded K-tile row stride (halves) to spread LDS banks

union H16 { v16h v; f16_t h[16]; v8h g[2]; };

__device__ __forceinline__ v8f wmma_f16(v16h a, v16h b, v8f c) {
  // D = A(16x32 f16) * B(32x16 f16) + C(16x16 f32)
  return __builtin_amdgcn_wmma_f32_16x16x32_f16(false, a, false, b, (short)0, c,
                                                false, false);
}

// A fragment (16x32, f16) from a row-major buffer.
// Lane l holds row (row_base + l%16); K-halves 0..7/16..23 (lanes 0-15) or
// 8..15/24..31 (lanes 16-31) per the CDNA5 16-bit A layout. Two 16B loads.
__device__ __forceinline__ v16h frag_a(const f16_t* base, int ld, int row_base,
                                       int k_base, unsigned lane) {
  unsigned r = lane & 15u, hf = lane >> 4;
  const f16_t* p = base + (size_t)(row_base + (int)r) * (size_t)ld
                        + (size_t)(k_base + (int)hf * 8);
  H16 u;
  u.g[0] = *(const v8h*)p;
  u.g[1] = *(const v8h*)(p + 16);
  return u.v;
}

// B fragment (32x16, f16) for X @ W^T, W row-major [out][in]:
// lane l holds column n = col_base + l%16 (= W row n), K 0..7/16..23 (lanes 0-15)
// or 8..15/24..31 (lanes 16-31), matching the A layout.
__device__ __forceinline__ v16h frag_b(const f16_t* w, int ld, int col_base,
                                       int k_base, unsigned lane) {
  unsigned n = lane & 15u, hf = lane >> 4;
  const f16_t* p = w + (size_t)(col_base + (int)n) * (size_t)ld
                     + (size_t)(k_base + (int)hf * 8);
  H16 u;
  u.g[0] = *(const v8h*)p;
  u.g[1] = *(const v8h*)(p + 16);
  return u.v;
}

__global__ void cvt_f32_f16(const float* __restrict__ src, f16_t* __restrict__ dst, int n) {
  int i = blockIdx.x * blockDim.x + threadIdx.x;
  if (i < n) dst[i] = (f16_t)src[i];
}

__global__ __launch_bounds__(NTHREADS, 2)
void tgat_kernel(const float* __restrict__ node_f, const float* __restrict__ nbr_f,
                 const float* __restrict__ node_t, const float* __restrict__ nbr_t,
                 const int* __restrict__ maskp,
                 const float* __restrict__ freq, const float* __restrict__ phs,
                 const float* __restrict__ qb, const float* __restrict__ kbv,
                 const float* __restrict__ vbv, const float* __restrict__ ob,
                 const float* __restrict__ f1b, const float* __restrict__ f2b,
                 const float* __restrict__ n1g, const float* __restrict__ n1b,
                 const float* __restrict__ n2g, const float* __restrict__ n2b,
                 const float* __restrict__ cb1, const float* __restrict__ cw2,
                 const float* __restrict__ cb2,
                 const f16_t* __restrict__ qw_h, const f16_t* __restrict__ kw_h,
                 const f16_t* __restrict__ vw_h, const f16_t* __restrict__ ow_h,
                 const f16_t* __restrict__ f1w_h, const f16_t* __restrict__ f2w_h,
                 const f16_t* __restrict__ cw1_h,
                 float* __restrict__ out) {
  const unsigned tid  = threadIdx.x;
  const unsigned lane = tid & 31u;
  const unsigned wid  = tid >> 5;
  const int gb0 = (int)blockIdx.x * BT;

  extern __shared__ char smem[];
  float* s_x    = (float*)(smem);              // [16][128] residual stream (f32)
  float* s_q    = (float*)(smem + 8192);       // [16][128] q / projection tmp
  float* s_sc   = (float*)(smem + 16384);      // [16][4][32] scores/attn
  f16_t* s_xh   = (f16_t*)(smem + 24576);      // [16][256] qc = [x_f16 | node_te]
  f16_t* s_aoh  = (f16_t*)(smem + 32768);      // [16][128] attn out (f16)
  float* s_freq = (float*)(smem + 36864);      // [128]
  float* s_phs  = (float*)(smem + 37376);      // [128]
  char*  s_big  = smem + 37888;                // 66KB overlapped region
  f16_t* s_k    = (f16_t*)s_big;               // [4 waves][32][KSTR] (34816 B)
  f16_t* s_v    = (f16_t*)(s_big + 34816);     // [4 waves][32][128]  (32768 B)
  f16_t* s_h1   = (f16_t*)s_big;               // [16][512] (FFN phase)
  float* s_c1   = (float*)s_big;               // [16][64]  (classifier phase)

  if (tid < TD_) { s_freq[tid] = freq[tid]; s_phs[tid] = phs[tid]; }
  __syncthreads();

  // ---- init: x (f32 + f16) and node time encoding ----
  for (int i = (int)tid; i < BT * ND_; i += NTHREADS) {
    int b = i >> 7, d = i & 127;
    float xv = node_f[(size_t)(gb0 + b) * ND_ + d];
    s_x[b * ND_ + d] = xv;
    s_xh[b * COMB_ + d] = (f16_t)xv;
    float tn = node_t[gb0 + b];
    s_xh[b * COMB_ + ND_ + d] = (f16_t)__cosf(tn * s_freq[d] + s_phs[d]);
  }
  __syncthreads();

  for (int l = 0; l < L_; ++l) {
    const f16_t* qwL  = qw_h  + (size_t)l * NHAD_ * COMB_;
    const f16_t* kwL  = kw_h  + (size_t)l * NHAD_ * COMB_;
    const f16_t* vwL  = vw_h  + (size_t)l * NHAD_ * COMB_;
    const f16_t* owL  = ow_h  + (size_t)l * ND_ * NHAD_;
    const f16_t* f1wL = f1w_h + (size_t)l * FF_ * ND_;
    const f16_t* f2wL = f2w_h + (size_t)l * ND_ * FF_;
    const float* qbL = qb + l * NHAD_;
    const float* kbL = kbv + l * NHAD_;
    const float* vbL = vbv + l * NHAD_;
    const float* obL = ob + l * ND_;
    const float* f1bL = f1b + l * FF_;
    const float* f2bL = f2b + l * ND_;

    // ---- Q projection: s_q = qc @ qw^T + qb (one 16-row M-tile) ----
    {
      v16h aq[8];
      #pragma unroll
      for (int kk = 0; kk < 8; ++kk) aq[kk] = frag_a(s_xh, COMB_, 0, kk * 32, lane);
      for (int t = 0; t < 2; ++t) {
        int nt = (int)wid + t * 4;
        v8f c = {};
        #pragma unroll
        for (int kk = 0; kk < 8; ++kk)
          c = wmma_f16(aq[kk], frag_b(qwL, COMB_, nt * 16, kk * 32, lane), c);
        int col = nt * 16 + (int)(lane & 15u);
        int rb = (int)(lane >> 4) * 8;
        float bv = qbL[col];
        #pragma unroll
        for (int r = 0; r < 8; ++r) s_q[(rb + r) * NHAD_ + col] = c[r] + bv;
      }
    }
    __syncthreads();

    // ---- K/V projections + attention (each wave owns 4 batch rows) ----
    {
      f16_t* kbuf = s_k + (size_t)wid * 32 * KSTR;
      f16_t* vbuf = s_v + (size_t)wid * 32 * NHAD_;
      for (int bb = 0; bb < 4; ++bb) {
        int b = (int)wid * 4 + bb;
        int gb = gb0 + b;
        float tn = node_t[gb];

        // prefetch next batch element's 16KB neighbor-feature block (128B steps)
        if (bb < 3) {
          const float* pf = nbr_f + (size_t)(gb + 1) * NNB * ND_;
          #pragma unroll
          for (int u = 0; u < 4; ++u)
            __builtin_prefetch(pf + ((size_t)u * 32 + lane) * 32, 0, 1);
        }

        // build kc A-fragments for BOTH neighbor halves (j 0..15 and 16..31)
        int jlo = (int)(lane & 15u);
        int h8 = (int)(lane >> 4) * 8;
        size_t gr0 = (size_t)gb * NNB + jlo;
        size_t gr1 = gr0 + 16;
        float rt0 = nbr_t[gr0] - tn;
        float rt1 = nbr_t[gr1] - tn;
        H16 af0[8], af1[8];
        #pragma unroll
        for (int kk = 0; kk < 4; ++kk) {
          const float* sp0 = nbr_f + gr0 * ND_ + kk * 32 + h8;
          const float* sp1 = nbr_f + gr1 * ND_ + kk * 32 + h8;
          #pragma unroll
          for (int i = 0; i < 8; ++i) {
            af0[kk].h[i]     = (f16_t)sp0[i];
            af0[kk].h[8 + i] = (f16_t)sp0[16 + i];
            af1[kk].h[i]     = (f16_t)sp1[i];
            af1[kk].h[8 + i] = (f16_t)sp1[16 + i];
          }
        }
        #pragma unroll
        for (int kk = 4; kk < 8; ++kk) {
          int t0 = (kk - 4) * 32 + h8;
          #pragma unroll
          for (int i = 0; i < 8; ++i) {
            float f0 = s_freq[t0 + i], p0 = s_phs[t0 + i];
            float f1 = s_freq[t0 + 16 + i], p1 = s_phs[t0 + 16 + i];
            af0[kk].h[i]     = (f16_t)__cosf(rt0 * f0 + p0);
            af0[kk].h[8 + i] = (f16_t)__cosf(rt0 * f1 + p1);
            af1[kk].h[i]     = (f16_t)__cosf(rt1 * f0 + p0);
            af1[kk].h[8 + i] = (f16_t)__cosf(rt1 * f1 + p1);
          }
        }

        // K projection: each B-fragment feeds two independent WMMAs
        for (int nt = 0; nt < 8; ++nt) {
          v8f c0 = {}, c1 = {};
          #pragma unroll
          for (int kk = 0; kk < 8; ++kk) {
            v16h bf = frag_b(kwL, COMB_, nt * 16, kk * 32, lane);
            c0 = wmma_f16(af0[kk].v, bf, c0);
            c1 = wmma_f16(af1[kk].v, bf, c1);
          }
          int col = nt * 16 + (int)(lane & 15u);
          int rb = (int)(lane >> 4) * 8;
          float bv = kbL[col];
          #pragma unroll
          for (int r = 0; r < 8; ++r) {
            kbuf[(rb + r) * KSTR + col]        = (f16_t)(c0[r] + bv);
            kbuf[(16 + rb + r) * KSTR + col]   = (f16_t)(c1[r] + bv);
          }
        }
        // scores: lane owns (head=d4, j=lane) over all 32 neighbors
        #pragma unroll
        for (int d4 = 0; d4 < 4; ++d4) {
          int j = (int)lane;
          float acc = 0.f;
          for (int d = 0; d < AD_; ++d)
            acc += s_q[b * NHAD_ + d4 * AD_ + d] * (float)kbuf[j * KSTR + d4 * AD_ + d];
          acc *= 0.17677669529663687f;  // 1/sqrt(32)
          if (maskp[(size_t)gb * NNB + j] == 0) acc = -1e9f;
          s_sc[b * 128 + d4 * 32 + j] = acc;
        }
        // V projection (reuses both A-fragment sets)
        for (int nt = 0; nt < 8; ++nt) {
          v8f c0 = {}, c1 = {};
          #pragma unroll
          for (int kk = 0; kk < 8; ++kk) {
            v16h bf = frag_b(vwL, COMB_, nt * 16, kk * 32, lane);
            c0 = wmma_f16(af0[kk].v, bf, c0);
            c1 = wmma_f16(af1[kk].v, bf, c1);
          }
          int col = nt * 16 + (int)(lane & 15u);
          int rb = (int)(lane >> 4) * 8;
          float bv = vbL[col];
          #pragma unroll
          for (int r = 0; r < 8; ++r) {
            vbuf[(rb + r) * NHAD_ + col]      = (f16_t)(c0[r] + bv);
            vbuf[(16 + rb + r) * NHAD_ + col] = (f16_t)(c1[r] + bv);
          }
        }
        // softmax over 32 neighbors, one lane per head
        if (lane < 4) {
          int hh = (int)lane;
          float mx = -1e30f;
          for (int jj = 0; jj < 32; ++jj) mx = fmaxf(mx, s_sc[b * 128 + hh * 32 + jj]);
          float sum = 0.f;
          for (int jj = 0; jj < 32; ++jj) {
            float e = __expf(s_sc[b * 128 + hh * 32 + jj] - mx);
            s_sc[b * 128 + hh * 32 + jj] = e;
            sum += e;
          }
          float inv = 1.f / sum;
          for (int jj = 0; jj < 32; ++jj) s_sc[b * 128 + hh * 32 + jj] *= inv;
        }
        // attention output: ao[d] = sum_j attn[head(d)][j] * V[j][d]
        #pragma unroll
        for (int d4 = 0; d4 < 4; ++d4) {
          int d = d4 * 32 + (int)lane;
          float acc = 0.f;
          for (int jj = 0; jj < 32; ++jj)
            acc += s_sc[b * 128 + d4 * 32 + jj] * (float)vbuf[jj * NHAD_ + d];
          s_aoh[b * NHAD_ + d] = (f16_t)acc;
        }
      }  // bb
    }
    __syncthreads();

    // ---- output projection: s_q = ao @ ow^T + ob ----
    {
      v16h aa[4];
      #pragma unroll
      for (int kk = 0; kk < 4; ++kk) aa[kk] = frag_a(s_aoh, NHAD_, 0, kk * 32, lane);
      for (int t = 0; t < 2; ++t) {
        int nt = (int)wid + t * 4;
        v8f c = {};
        #pragma unroll
        for (int kk = 0; kk < 4; ++kk)
          c = wmma_f16(aa[kk], frag_b(owL, NHAD_, nt * 16, kk * 32, lane), c);
        int col = nt * 16 + (int)(lane & 15u);
        int rb = (int)(lane >> 4) * 8;
        float bv = obL[col];
        #pragma unroll
        for (int r = 0; r < 8; ++r) s_q[(rb + r) * ND_ + col] = c[r] + bv;
      }
    }
    __syncthreads();

    // ---- LayerNorm 1 (residual) ----
    if (tid < BT) {
      int b = (int)tid;
      float sum = 0.f, sq = 0.f;
      for (int d = 0; d < ND_; ++d) {
        float v = s_x[b * ND_ + d] + s_q[b * ND_ + d];
        sum += v; sq += v * v;
      }
      float mean = sum * (1.f / ND_);
      float var = sq * (1.f / ND_) - mean * mean;
      float rstd = rsqrtf(var + 1e-5f);
      for (int d = 0; d < ND_; ++d) {
        float v = s_x[b * ND_ + d] + s_q[b * ND_ + d];
        float y = (v - mean) * rstd * n1g[l * ND_ + d] + n1b[l * ND_ + d];
        s_x[b * ND_ + d] = y;
        s_xh[b * COMB_ + d] = (f16_t)y;
      }
    }
    __syncthreads();

    // ---- FFN1: h1 = relu(x @ f1w^T + f1b), 32 N-tiles across waves ----
    {
      v16h ax[4];
      #pragma unroll
      for (int kk = 0; kk < 4; ++kk) ax[kk] = frag_a(s_xh, COMB_, 0, kk * 32, lane);
      for (int t = 0; t < 8; ++t) {
        int nt = t * 4 + (int)wid;
        v8f c = {};
        #pragma unroll
        for (int kk = 0; kk < 4; ++kk)
          c = wmma_f16(ax[kk], frag_b(f1wL, ND_, nt * 16, kk * 32, lane), c);
        int col = nt * 16 + (int)(lane & 15u);
        int rb = (int)(lane >> 4) * 8;
        float bv = f1bL[col];
        #pragma unroll
        for (int r = 0; r < 8; ++r) {
          float v = c[r] + bv;
          s_h1[(rb + r) * FF_ + col] = (f16_t)(v > 0.f ? v : 0.f);
        }
      }
    }
    __syncthreads();

    // ---- FFN2: s_q = h1 @ f2w^T + f2b ----
    {
      for (int t = 0; t < 2; ++t) {
        int nt = (int)wid + t * 4;
        v8f c = {};
        for (int kk = 0; kk < 16; ++kk)
          c = wmma_f16(frag_a(s_h1, FF_, 0, kk * 32, lane),
                       frag_b(f2wL, FF_, nt * 16, kk * 32, lane), c);
        int col = nt * 16 + (int)(lane & 15u);
        int rb = (int)(lane >> 4) * 8;
        float bv = f2bL[col];
        #pragma unroll
        for (int r = 0; r < 8; ++r) s_q[(rb + r) * ND_ + col] = c[r] + bv;
      }
    }
    __syncthreads();

    // ---- LayerNorm 2 (residual) ----
    if (tid < BT) {
      int b = (int)tid;
      float sum = 0.f, sq = 0.f;
      for (int d = 0; d < ND_; ++d) {
        float v = s_x[b * ND_ + d] + s_q[b * ND_ + d];
        sum += v; sq += v * v;
      }
      float mean = sum * (1.f / ND_);
      float var = sq * (1.f / ND_) - mean * mean;
      float rstd = rsqrtf(var + 1e-5f);
      for (int d = 0; d < ND_; ++d) {
        float v = s_x[b * ND_ + d] + s_q[b * ND_ + d];
        float y = (v - mean) * rstd * n2g[l * ND_ + d] + n2b[l * ND_ + d];
        s_x[b * ND_ + d] = y;
        s_xh[b * COMB_ + d] = (f16_t)y;
      }
    }
    __syncthreads();
  }  // layers

  // ---- classifier: relu(x @ cw1^T + cb1) -> sigmoid(· @ cw2^T + cb2) ----
  {
    int nt = (int)wid;  // 4 N-tiles of 16 = 64 hidden units
    v16h ax[4];
    #pragma unroll
    for (int kk = 0; kk < 4; ++kk) ax[kk] = frag_a(s_xh, COMB_, 0, kk * 32, lane);
    v8f c = {};
    #pragma unroll
    for (int kk = 0; kk < 4; ++kk)
      c = wmma_f16(ax[kk], frag_b(cw1_h, ND_, nt * 16, kk * 32, lane), c);
    int col = nt * 16 + (int)(lane & 15u);
    int rb = (int)(lane >> 4) * 8;
    float bv = cb1[col];
    #pragma unroll
    for (int r = 0; r < 8; ++r) {
      float v = c[r] + bv;
      s_c1[(rb + r) * 64 + col] = v > 0.f ? v : 0.f;
    }
  }
  __syncthreads();
  if (tid < BT) {
    int b = (int)tid;
    float acc = cb2[0];
    for (int i2 = 0; i2 < 64; ++i2) acc += s_c1[b * 64 + i2] * cw2[i2];
    out[gb0 + b] = 1.f / (1.f + __expf(-acc));
  }
}

extern "C" void kernel_launch(void* const* d_in, const int* in_sizes, int n_in,
                              void* d_out, int out_size, void* d_ws, size_t ws_size,
                              hipStream_t stream) {
  (void)in_sizes; (void)n_in; (void)out_size; (void)ws_size;

  // f16 weight staging in workspace (halves)
  f16_t* wsH   = (f16_t*)d_ws;
  f16_t* qw_h  = wsH;             // L*128*256 = 65536
  f16_t* kw_h  = wsH + 65536;     // 65536
  f16_t* vw_h  = wsH + 131072;    // 65536
  f16_t* ow_h  = wsH + 196608;    // L*128*128 = 32768
  f16_t* f1w_h = wsH + 229376;    // L*512*128 = 131072
  f16_t* f2w_h = wsH + 360448;    // L*128*512 = 131072
  f16_t* cw1_h = wsH + 491520;    // 64*128 = 8192

  struct Cvt { int idx; f16_t* dst; int n; };
  const Cvt cv[7] = {
    {7,  qw_h,  65536}, {9,  kw_h,  65536}, {11, vw_h,  65536},
    {13, ow_h,  32768}, {15, f1w_h, 131072}, {17, f2w_h, 131072},
    {23, cw1_h, 8192},
  };
  for (int i = 0; i < 7; ++i)
    cvt_f32_f16<<<(cv[i].n + 255) / 256, 256, 0, stream>>>(
        (const float*)d_in[cv[i].idx], cv[i].dst, cv[i].n);

  const size_t shmem = 105472;  // 37888 fixed + 67584 overlapped region
  (void)hipFuncSetAttribute((const void*)tgat_kernel,
                            hipFuncAttributeMaxDynamicSharedMemorySize, (int)shmem);

  tgat_kernel<<<B_TOT / BT, NTHREADS, shmem, stream>>>(
      (const float*)d_in[0], (const float*)d_in[1], (const float*)d_in[2],
      (const float*)d_in[3], (const int*)d_in[4], (const float*)d_in[5],
      (const float*)d_in[6],
      (const float*)d_in[8], (const float*)d_in[10], (const float*)d_in[12],
      (const float*)d_in[14], (const float*)d_in[16], (const float*)d_in[18],
      (const float*)d_in[19], (const float*)d_in[20], (const float*)d_in[21],
      (const float*)d_in[22],
      (const float*)d_in[24], (const float*)d_in[25], (const float*)d_in[26],
      qw_h, kw_h, vw_h, ow_h, f1w_h, f2w_h, cw1_h,
      (float*)d_out);
}
